// BasicBlock_35545149341763
// MI455X (gfx1250) — 8-bit matrix operand order repaired and checked by reading, compiled, not run on hardware
//
#include <hip/hip_runtime.h>
#include <cstdint>

typedef __attribute__((ext_vector_type(8))) int v8i;

#define ALPHA_F 0.2f
#define BETA_F  1.0f
#define EPS_W   1e-5f

#define N_IMG 32
#define CIN   512
#define COUT  512
#define HH    28
#define WW    28
#define HP    30
#define WP    30
#define FANIN 4608
#define NPIX  (N_IMG*HH*WW)
#define BM    128
#define BN    128
#define BK    128
#define MT    (NPIX/BM)
#define NT    (COUT/BN)
#define NSTEP (9 * (CIN/BK))

#define BACT_BYTES (N_IMG*HP*WP*CIN)
#define WBIN_BYTES (9*COUT*CIN)

__global__ __launch_bounds__(256) void wprep_kernel(
    const float* __restrict__ w, const float* __restrict__ gain,
    signed char* __restrict__ wbin, float* __restrict__ scl)
{
    __shared__ float redA[256];
    __shared__ float redB[256];
    const int co  = blockIdx.x;
    const int tid = threadIdx.x;
    const float* wc = w + co * FANIN;

    float s = 0.f, s2 = 0.f;
    for (int i = tid; i < FANIN; i += 256) { float v = wc[i]; s += v; s2 += v * v; }
    redA[tid] = s; redB[tid] = s2; __syncthreads();
    for (int off = 128; off > 0; off >>= 1) {
        if (tid < off) { redA[tid] += redA[tid + off]; redB[tid] += redB[tid + off]; }
        __syncthreads();
    }
    const float mean = redA[0] * (1.f / FANIN);
    const float var0 = fmaxf(redB[0] * (1.f / FANIN) - mean * mean, 0.f);
    const float stdv = sqrtf(var0);
    __syncthreads();

    float sa = 0.f;
    for (int i = tid; i < FANIN; i += 256) sa += fabsf(wc[i] - mean);
    redA[tid] = sa; __syncthreads();
    for (int off = 128; off > 0; off >>= 1) {
        if (tid < off) redA[tid] += redA[tid + off];
        __syncthreads();
    }
    const float ws_scale = 1.0f / sqrtf((float)FANIN);
    const float inv      = ws_scale / (stdv + EPS_W);
    const float sf       = inv * redA[0] * (1.f / FANIN);

    for (int i = tid; i < FANIN; i += 256) {
        const int ci = i / 9;
        const int t  = i - ci * 9;
        const float d = wc[i] - mean;
        const signed char sg = (d > 0.f) ? 1 : ((d < 0.f) ? -1 : 0);
        wbin[(t * COUT + co) * CIN + ci] = sg;
    }
    if (tid == 0) scl[co] = sf * gain[co] * ALPHA_F;
}

__global__ __launch_bounds__(256) void zero_border_kernel(signed char* __restrict__ bact)
{
    const int gid  = blockIdx.x * 256 + threadIdx.x;
    const int t16  = gid & 31;
    const int cell = (gid >> 5) % 116;
    const int n    = (gid >> 5) / 116;
    int hp, wp;
    if (cell < 30)      { hp = 0;  wp = cell; }
    else if (cell < 60) { hp = 29; wp = cell - 30; }
    else if (cell < 88) { hp = cell - 60 + 1; wp = 0; }
    else                { hp = cell - 88 + 1; wp = 29; }
    const int4 z = {0, 0, 0, 0};
    *(int4*)&bact[(((n * HP + hp) * WP + wp) << 9) + t16 * 16] = z;
}

__global__ __launch_bounds__(256) void aprep_kernel(
    const float* __restrict__ x, const float* __restrict__ b0,
    signed char* __restrict__ bact)
{
    __shared__ signed char tile[HH * WW * 16];
    const int n   = blockIdx.x >> 5;
    const int g   = blockIdx.x & 31;
    const int c0  = g * 16;
    const int tid = threadIdx.x;

    #pragma unroll 1
    for (int cl = 0; cl < 16; ++cl) {
        const float b = b0[c0 + cl];
        const float* xc = x + (n * CIN + c0 + cl) * (HH * WW);
        for (int hw = tid; hw < HH * WW; hw += 256) {
            const float v = xc[hw] * BETA_F + b;
            tile[hw * 16 + cl] = (v > 0.f) ? 1 : ((v < 0.f) ? -1 : 0);
        }
    }
    __syncthreads();
    for (int hw = tid; hw < HH * WW; hw += 256) {
        const int h = hw / WW;
        const int w = hw - h * WW;
        *(int4*)&bact[(((n * HP + h + 1) * WP + (w + 1)) << 9) + c0] =
            *(const int4*)&tile[hw * 16];
    }
}

__device__ __forceinline__ void async_copy_b128(uint32_t lds_addr, uint32_t goff,
                                                const signed char* base)
{
    asm volatile("global_load_async_to_lds_b128 %0, %1, %2"
                 :: "v"(lds_addr), "v"(goff),
                    "s"((unsigned long long)(uintptr_t)base)
                 : "memory");
}

__global__ __launch_bounds__(256) void bgemm_kernel(
    const signed char* __restrict__ bact, const signed char* __restrict__ wbin,
    const float* __restrict__ scl, const float* __restrict__ x,
    const float* __restrict__ move1, const float* __restrict__ pa,
    const float* __restrict__ move2, float* __restrict__ out)
{
    __shared__ __align__(16) signed char lA[2][BM * BK];
    __shared__ __align__(16) signed char lB[2][BN * BK];

    const int tid  = threadIdx.x;
    const int lane = tid & 31;
    const int wid  = tid >> 5;
    const int mt   = blockIdx.x % MT;
    const int nt   = blockIdx.x / MT;
    const int m_base = mt * BM;
    const int n_base = nt * BN;
    const int wm = (wid & 3) * 32;
    const int wn = (wid >> 2) * 64;

    const int seg = (tid & 7) * 16;
    uint32_t a_goff[4], b_goff[4], a_lds[4], b_lds[4];
    const uint32_t lA0 = (uint32_t)(uintptr_t)&lA[0][0];
    const uint32_t lB0 = (uint32_t)(uintptr_t)&lB[0][0];
    #pragma unroll
    for (int i = 0; i < 4; ++i) {
        const int row = (tid >> 3) + i * 32;
        const int p   = m_base + row;
        const int n   = p / (HH * WW);
        const int hw  = p - n * (HH * WW);
        const int h   = hw / WW;
        const int w   = hw - h * WW;
        a_goff[i] = (uint32_t)((((n * HP + h) * WP + w) << 9) + seg);
        a_lds[i]  = lA0 + (uint32_t)(row * BK + seg);
        b_goff[i] = (uint32_t)(((n_base + row) << 9) + seg);
        b_lds[i]  = lB0 + (uint32_t)(row * BK + seg);
    }

    auto stage = [&](int s, int buf) {
        const int t = s >> 2;
        const int q = s & 3;
        const uint32_t offA = (uint32_t)((((t / 3) * WP + (t % 3)) << 9) + (q << 7));
        const uint32_t offB = (uint32_t)(t * (COUT * CIN) + (q << 7));
        const uint32_t lofs = (uint32_t)(buf * (BM * BK));
        if (t < 8) __builtin_prefetch(wbin + (t + 1) * (COUT * CIN), 0, 0);
        #pragma unroll
        for (int i = 0; i < 4; ++i) {
            async_copy_b128(a_lds[i] + lofs, a_goff[i] + offA, bact);
            async_copy_b128(b_lds[i] + lofs, b_goff[i] + offB, wbin);
        }
    };

    v8i acc[2][4];
    #pragma unroll
    for (int i = 0; i < 2; ++i)
        #pragma unroll
        for (int j = 0; j < 4; ++j)
            acc[i][j] = (v8i){0, 0, 0, 0, 0, 0, 0, 0};

    const int hsel = lane >> 4;
    const int lrow = lane & 15;

    stage(0, 0);
    #pragma unroll 1
    for (int s = 0; s < NSTEP; ++s) {
        const int buf = s & 1;
        if (s + 1 < NSTEP) {
            stage(s + 1, buf ^ 1);
            asm volatile("s_wait_asynccnt 8" ::: "memory");
        } else {
            asm volatile("s_wait_asynccnt 0" ::: "memory");
        }
        __syncthreads();

        #pragma unroll
        for (int ka = 0; ka < BK; ka += 64) {
            union { v8i v; long long l[4]; } af[2], bf;
            #pragma unroll
            for (int fm = 0; fm < 2; ++fm) {
                const int ml = wm + fm * 16 + lrow;
                #pragma unroll
                for (int vp = 0; vp < 4; ++vp)
                    af[fm].l[vp] = *(const long long*)
                        &lA[buf][ml * BK + ka + vp * 16 + hsel * 8];
            }
            #pragma unroll
            for (int fn = 0; fn < 4; ++fn) {
                const int nl = wn + fn * 16 + lrow;
                #pragma unroll
                for (int vp = 0; vp < 4; ++vp)
                    bf.l[vp] = *(const long long*)
                        &lB[buf][nl * BK + ka + vp * 16 + hsel * 8];
                #pragma unroll
                for (int fm = 0; fm < 2; ++fm)
                    acc[fm][fn] = __builtin_amdgcn_wmma_i32_16x16x64_iu8(
                         true, af[fm].v,  true, bf.v,
                        acc[fm][fn],  false,  false);
            }
        }
        __syncthreads();
    }

    #pragma unroll
    for (int fn = 0; fn < 4; ++fn) {
        const int co = n_base + wn + fn * 16 + lrow;
        const float sc = scl[co];
        const float m1 = move1[co];
        const float a  = pa[co];
        const float m2 = move2[co];
        #pragma unroll
        for (int fm = 0; fm < 2; ++fm) {
            union { v8i v; int i[8]; } u; u.v = acc[fm][fn];
            const int p0  = m_base + wm + fm * 16 + hsel * 8;
            const int n   = p0 / (HH * WW);
            const int hw0 = p0 - n * (HH * WW);
            const int idx = (n * COUT + co) * (HH * WW) + hw0;
            const float4 x0 = *(const float4*)&x[idx];
            const float4 x1 = *(const float4*)&x[idx + 4];
            float r[8];
            #pragma unroll
            for (int k = 0; k < 8; ++k) {
                const float xr = (k < 4) ? ((const float*)&x0)[k]
                                         : ((const float*)&x1)[k - 4];
                float v = (float)u.i[k] * sc + xr + m1;
                v = (v >= 0.f) ? v : a * v;
                r[k] = v + m2;
            }
            float4 o0 = {r[0], r[1], r[2], r[3]};
            float4 o1 = {r[4], r[5], r[6], r[7]};
            *(float4*)&out[idx]     = o0;
            *(float4*)&out[idx + 4] = o1;
        }
    }
}

extern "C" void kernel_launch(void* const* d_in, const int* in_sizes, int n_in,
                              void* d_out, int out_size, void* d_ws, size_t ws_size,
                              hipStream_t stream)
{
    const float* x     = (const float*)d_in[0];
    const float* b0    = (const float*)d_in[1];
    const float* wconv = (const float*)d_in[2];
    const float* gain  = (const float*)d_in[3];
    const float* move1 = (const float*)d_in[4];
    const float* pa    = (const float*)d_in[5];
    const float* move2 = (const float*)d_in[6];
    float* out = (float*)d_out;

    signed char* bact = (signed char*)d_ws;
    signed char* wbin = bact + BACT_BYTES;
    float*       scl  = (float*)(wbin + WBIN_BYTES);

    wprep_kernel<<<COUT, 256, 0, stream>>>(wconv, gain, wbin, scl);
    zero_border_kernel<<<(N_IMG * 116 * 32) / 256, 256, 0, stream>>>(bact);
    aprep_kernel<<<N_IMG * 32, 256, 0, stream>>>(x, b0, bact);
    bgemm_kernel<<<MT * NT, 256, 0, stream>>>(bact, wbin, scl, x, move1, pa, move2, out);
}
